// GATConv_68289980007158
// MI455X (gfx1250) — matrix operand order repaired and checked by reading, not compiled, not run on hardware
//
#include <hip/hip_runtime.h>
#include <hip/hip_bf16.h>

typedef __attribute__((ext_vector_type(16))) _Float16 v16h;
typedef __attribute__((ext_vector_type(8)))  _Float16 v8h;
typedef __attribute__((ext_vector_type(8)))  float    v8f;

#define D_IN      128
#define D_OUT     128
#define NHEAD     4
#define CHEAD     32
#define GEMM_ROWS 64        // rows per block (4 M-tiles)

// ---- monotone float<->uint encoding for atomicMax on floats -------------
__device__ __forceinline__ unsigned fenc(float f) {
    unsigned u = __float_as_uint(f);
    return (u & 0x80000000u) ? ~u : (u | 0x80000000u);
}
__device__ __forceinline__ float fdec(unsigned u) {
    return (u & 0x80000000u) ? __uint_as_float(u & 0x7fffffffu)
                             : __uint_as_float(~u);
}

// =========================================================================
// Kernel 1: h = x @ W + b_lin   via v_wmma_f32_16x16x32_f16
// Block = 256 threads = 8 waves, covers 64 rows x all 128 cols.
// Wave w owns column tile [16w,16w+16) and iterates 4 M-tiles, reusing its
// B fragment across them. W is staged TRANSPOSED in LDS so B fragments are
// contiguous 16-f16 runs -> 2x ds_load_b128 each (no scalar gathers).
// =========================================================================
__global__ __launch_bounds__(256) void gat_gemm_wmma(
    const float* __restrict__ x, const float* __restrict__ W,
    const float* __restrict__ b_lin, float* __restrict__ h, int Nn)
{
    __shared__ _Float16 ldsWt[D_OUT * D_IN];        // transposed [col][k], 32 KB
    __shared__ _Float16 ldsX[GEMM_ROWS * D_IN];     // [row][k], 16 KB

    const int tid  = threadIdx.x;
    const int row0 = blockIdx.x * GEMM_ROWS;

    // Stage W (fp32 -> f16, transposed). W stays hot in the 192 MB L2.
    for (int idx = tid; idx < D_IN * D_OUT; idx += 256) {
        int k = idx >> 7, n = idx & 127;
        ldsWt[n * D_IN + k] = (_Float16)W[idx];
    }
    // Stage the 64x128 x-tile (zero-fill past Nn).
    for (int idx = tid; idx < GEMM_ROWS * D_IN; idx += 256) {
        int r = idx >> 7, c = idx & 127;
        int gr = row0 + r;
        ldsX[idx] = (_Float16)((gr < Nn) ? x[(size_t)gr * D_IN + c] : 0.0f);
    }
    __syncthreads();

    const int wave  = tid >> 5;          // 0..7 -> N tile
    const int lane  = tid & 31;
    const int m     = lane & 15;         // row (A) / col (B,C) within tile
    const int khalf = lane >> 4;         // which K half this lane holds
    const int col0  = wave * 16;

    v8f acc[4] = {};                     // 4 M-tiles
    #pragma unroll
    for (int ks = 0; ks < 4; ++ks) {     // K = 128 in 4 steps of 32
        // B 32x16 f16 layout: elem e -> K = 16*khalf + e (contiguous run)
        v16h bfrag;
        {
            const _Float16* bp = &ldsWt[(col0 + m) * D_IN + ks * 32 + khalf * 16];
            *((v8h*)&bfrag)     = *(const v8h*)(bp);
            *((v8h*)&bfrag + 1) = *(const v8h*)(bp + 8);
        }
        #pragma unroll
        for (int t = 0; t < 4; ++t) {
            // A 16x32 f16 layout: elem e -> K = 16*khalf + e (contiguous run)
            v16h afrag;
            const _Float16* ap = &ldsX[(t * 16 + m) * D_IN + ks * 32 + khalf * 16];
            *((v8h*)&afrag)     = *(const v8h*)(ap);
            *((v8h*)&afrag + 1) = *(const v8h*)(ap + 8);
            acc[t] = __builtin_amdgcn_wmma_f32_16x16x32_f16(
                         false, afrag, false, bfrag, (short)0, acc[t], false, false);
        }
    }

    // C layout: VGPR r -> M = 8*khalf + r, N = lane%16
    #pragma unroll
    for (int t = 0; t < 4; ++t) {
        #pragma unroll
        for (int r = 0; r < 8; ++r) {
            int gm = row0 + t * 16 + khalf * 8 + r;
            int gc = col0 + m;
            if (gm < Nn)
                h[(size_t)gm * D_OUT + gc] = acc[t][r] + b_lin[gc];
        }
    }
}

// =========================================================================
// Kernel 2: per-node attention scores  ai[n,h] = h[n,h,:]·att_i[h,:], aj likewise
// =========================================================================
__global__ void gat_att_scores(const float* __restrict__ h,
                               const float* __restrict__ att,
                               float* __restrict__ ai, float* __restrict__ aj,
                               int Nn)
{
    int t = blockIdx.x * blockDim.x + threadIdx.x;   // t = n*NHEAD + head
    if (t >= Nn * NHEAD) return;
    int node = t >> 2, head = t & 3;
    const float4* hp = (const float4*)(h + (size_t)node * D_OUT + head * CHEAD);
    const float4* pi = (const float4*)(att + head * (2 * CHEAD));
    const float4* pj = pi + (CHEAD / 4);
    float si = 0.f, sj = 0.f;
    #pragma unroll
    for (int c = 0; c < CHEAD / 4; ++c) {
        float4 v = hp[c], a = pi[c], b = pj[c];
        si += v.x * a.x + v.y * a.y + v.z * a.z + v.w * a.w;
        sj += v.x * b.x + v.y * b.y + v.z * b.z + v.w * b.w;
    }
    ai[t] = si; aj[t] = sj;
}

// =========================================================================
// Kernel 3: init segmax/-inf, segsum/0, out/0
// =========================================================================
__global__ void gat_init(unsigned* __restrict__ segmax, float* __restrict__ segsum,
                         float* __restrict__ out, int nh, int nout)
{
    int t = blockIdx.x * blockDim.x + threadIdx.x;
    if (t < nout) out[t] = 0.0f;
    if (t < nh) { segmax[t] = fenc(-3.0e38f); segsum[t] = 0.0f; }
}

__device__ __forceinline__ float leaky(float l) { return l > 0.0f ? l : 0.2f * l; }

// =========================================================================
// Kernel 4: segment max of leaky-relu logits (one thread per edge, 4 heads)
// =========================================================================
__global__ void gat_pass_max(const long long* __restrict__ ei,
                             const float* __restrict__ ai, const float* __restrict__ aj,
                             unsigned* __restrict__ segmax, int E, int T)
{
    int e = blockIdx.x * blockDim.x + threadIdx.x;
    if (e >= T) return;
    int i, j;
    if (e < E) { i = (int)ei[e]; j = (int)ei[E + e]; } else { i = j = e - E; }
    float4 av = ((const float4*)ai)[i];
    float4 jv = ((const float4*)aj)[j];
    atomicMax(&segmax[i * 4 + 0], fenc(leaky(av.x + jv.x)));
    atomicMax(&segmax[i * 4 + 1], fenc(leaky(av.y + jv.y)));
    atomicMax(&segmax[i * 4 + 2], fenc(leaky(av.z + jv.z)));
    atomicMax(&segmax[i * 4 + 3], fenc(leaky(av.w + jv.w)));
}

// =========================================================================
// Kernel 5: segment sum of exp(logit - segmax)
// =========================================================================
__global__ void gat_pass_sum(const long long* __restrict__ ei,
                             const float* __restrict__ ai, const float* __restrict__ aj,
                             const unsigned* __restrict__ segmax,
                             float* __restrict__ segsum, int E, int T)
{
    int e = blockIdx.x * blockDim.x + threadIdx.x;
    if (e >= T) return;
    int i, j;
    if (e < E) { i = (int)ei[e]; j = (int)ei[E + e]; } else { i = j = e - E; }
    float4 av = ((const float4*)ai)[i];
    float4 jv = ((const float4*)aj)[j];
    uint4  mx = ((const uint4*)segmax)[i];
    atomicAdd(&segsum[i * 4 + 0], __expf(leaky(av.x + jv.x) - fdec(mx.x)));
    atomicAdd(&segsum[i * 4 + 1], __expf(leaky(av.y + jv.y) - fdec(mx.y)));
    atomicAdd(&segsum[i * 4 + 2], __expf(leaky(av.z + jv.z) - fdec(mx.z)));
    atomicAdd(&segsum[i * 4 + 3], __expf(leaky(av.w + jv.w) - fdec(mx.w)));
}

// =========================================================================
// Kernel 6: out[i] += alpha * h[j]   (one wave32 per edge; 4 x 32 channels)
// =========================================================================
__global__ __launch_bounds__(256) void gat_aggregate(
    const long long* __restrict__ ei,
    const float* __restrict__ ai, const float* __restrict__ aj,
    const unsigned* __restrict__ segmax, const float* __restrict__ segsum,
    const float* __restrict__ h, float* __restrict__ out, int E, int T)
{
    int gtid = blockIdx.x * blockDim.x + threadIdx.x;
    int wid  = gtid >> 5;        // edge id
    int lane = threadIdx.x & 31;
    if (wid >= T) return;
    // stream-ahead prefetch of the edge list (global_prefetch_b8)
    if (wid + 4096 < E && lane == 0) {
        __builtin_prefetch(&ei[wid + 4096], 0, 1);
        __builtin_prefetch(&ei[E + wid + 4096], 0, 1);
    }
    int i, j;
    if (wid < E) { i = (int)ei[wid]; j = (int)ei[E + wid]; } else { i = j = wid - E; }

    float4 av = ((const float4*)ai)[i];
    float4 jv = ((const float4*)aj)[j];
    uint4  mx = ((const uint4*)segmax)[i];
    float4 sm = ((const float4*)segsum)[i];
    float alpha[NHEAD];
    alpha[0] = __expf(leaky(av.x + jv.x) - fdec(mx.x)) / sm.x;
    alpha[1] = __expf(leaky(av.y + jv.y) - fdec(mx.y)) / sm.y;
    alpha[2] = __expf(leaky(av.z + jv.z) - fdec(mx.z)) / sm.z;
    alpha[3] = __expf(leaky(av.w + jv.w) - fdec(mx.w)) / sm.w;

    const float* hj = h + (size_t)j * D_OUT;
    float* oi = out + (size_t)i * D_OUT;
    #pragma unroll
    for (int c0 = 0; c0 < NHEAD; ++c0) {
        int c = c0 * 32 + lane;            // head = c/32 = c0
        atomicAdd(&oi[c], alpha[c0] * hj[c]);
    }
}

// =========================================================================
// Kernel 7: out = relu(out + bias)   (in place, elementwise)
// =========================================================================
__global__ void gat_finalize(float* __restrict__ out, const float* __restrict__ bias, int n)
{
    int t = blockIdx.x * blockDim.x + threadIdx.x;
    if (t >= n) return;
    float v = out[t] + bias[t & (D_OUT - 1)];
    out[t] = v > 0.0f ? v : 0.0f;
}

// =========================================================================
extern "C" void kernel_launch(void* const* d_in, const int* in_sizes, int n_in,
                              void* d_out, int out_size, void* d_ws, size_t ws_size,
                              hipStream_t stream)
{
    const float*     x     = (const float*)d_in[0];
    const long long* ei    = (const long long*)d_in[1];   // int64 (2,E)
    const float*     W     = (const float*)d_in[2];
    const float*     b_lin = (const float*)d_in[3];
    const float*     att   = (const float*)d_in[4];
    const float*     bias  = (const float*)d_in[5];
    float*           out   = (float*)d_out;

    const int Nn = in_sizes[0] / D_IN;      // 50000
    const int E  = in_sizes[1] / 2;         // 800000
    const int T  = E + Nn;                  // edges + self loops

    // workspace layout (~29 MB)
    char* ws = (char*)d_ws;
    float*    h      = (float*)ws;                               // Nn*128
    float*    ai     = (float*)(ws + (size_t)Nn * D_OUT * 4);    // Nn*4
    float*    aj     = ai + (size_t)Nn * NHEAD;                  // Nn*4
    unsigned* segmax = (unsigned*)(aj + (size_t)Nn * NHEAD);     // Nn*4
    float*    segsum = (float*)(segmax + (size_t)Nn * NHEAD);    // Nn*4

    // 1) h = x@W + b  (WMMA)
    gat_gemm_wmma<<<(Nn + GEMM_ROWS - 1) / GEMM_ROWS, 256, 0, stream>>>(x, W, b_lin, h, Nn);
    // 2) per-node attention scores
    gat_att_scores<<<(Nn * NHEAD + 255) / 256, 256, 0, stream>>>(h, att, ai, aj, Nn);
    // 3) init accumulators + output
    gat_init<<<(Nn * D_OUT + 255) / 256, 256, 0, stream>>>(segmax, segsum, out,
                                                           Nn * NHEAD, Nn * D_OUT);
    // 4) segment max
    gat_pass_max<<<(T + 255) / 256, 256, 0, stream>>>(ei, ai, aj, segmax, E, T);
    // 5) segment sum of exp
    gat_pass_sum<<<(T + 255) / 256, 256, 0, stream>>>(ei, ai, aj, segmax, segsum, E, T);
    // 6) weighted aggregation (wave per edge)
    gat_aggregate<<<((size_t)T * 32 + 255) / 256, 256, 0, stream>>>(
        ei, ai, aj, segmax, segsum, h, out, E, T);
    // 7) bias + relu
    gat_finalize<<<(Nn * D_OUT + 255) / 256, 256, 0, stream>>>(out, bias, Nn * D_OUT);
}
